// ContrastiveLoss_36283883717202
// MI455X (gfx1250) — compile-verified, matrix operand order checked by reading, not run on hardware
//
#include <hip/hip_runtime.h>
#include <hip/hip_bf16.h>

typedef __attribute__((ext_vector_type(16))) _Float16 v16h;
typedef __attribute__((ext_vector_type(8)))  _Float16 v8h;
typedef __attribute__((ext_vector_type(8)))  float    v8f;

#define N_       4096
#define D_       128
#define INV_TEMP 20.0f   // 1 / 0.05
#define LDS_PITCH 136    // 128 halves + 8 pad -> 272B row stride, kills bank conflicts

// ---- ordered-uint encoding of float for integer atomicMin --------------------
__device__ __forceinline__ unsigned fkey(float f) {
    unsigned u = __float_as_uint(f);
    return (u & 0x80000000u) ? ~u : (u | 0x80000000u);
}
__device__ __forceinline__ float fdec(unsigned k) {
    return __uint_as_float((k & 0x80000000u) ? (k & 0x7fffffffu) : ~k);
}

// ---- init: zero accumulators + output, min-keys to max ----------------------
__global__ void init_kernel(unsigned* __restrict__ minkey, float* __restrict__ den_y,
                            float* __restrict__ den_x, float* __restrict__ out) {
    int t = blockIdx.x * blockDim.x + threadIdx.x;
    if (t < N_) {
        minkey[t] = 0xFFFFFFFFu;
        den_y[t]  = 0.0f;
        den_x[t]  = 0.0f;
    }
    if (t == 0) out[0] = 0.0f;
}

// ---- f32 -> f16 conversion ---------------------------------------------------
__global__ void cvt_kernel(const float* __restrict__ src, _Float16* __restrict__ dst, int n) {
    int t = blockIdx.x * blockDim.x + threadIdx.x;
    if (t < n) dst[t] = (_Float16)src[t];
}

// ---- fused WMMA GEMM + masked exp/min epilogue ------------------------------
// Block = 8 waves computes a 128(M) x 64(N) macro-tile.
// B strip (64 cols x 128 K, f16) staged in LDS via async global->LDS loads and
// shared by all waves; each wave owns 16 rows and 4 accumulators (16x64).
// mode 0: B = yf, col track = j>>3, own cols -> atomicMin(sim_p), others -> den_y
// mode 1: B = x,  col track = tr[j], same-track -> excluded, others -> den_x
__global__ __launch_bounds__(256) void sim_kernel(const _Float16* __restrict__ A,
                                                  const _Float16* __restrict__ B,
                                                  const int* __restrict__ tr,
                                                  unsigned* __restrict__ minkey,
                                                  float* __restrict__ den,
                                                  int mode) {
    __shared__ _Float16 sB[64 * LDS_PITCH];   // 17408 B

    const int wave  = threadIdx.x >> 5;
    const int lane  = threadIdx.x & 31;
    const int lrow  = lane & 15;
    const int half_ = lane >> 4;
    const int kbase = half_ * 8;              // K sub-select per half-wave (ISA layout)

    const int tileM   = blockIdx.x >> 6;      // 32 macro row-tiles of 128
    const int tileN6  = blockIdx.x & 63;      // 64 macro col-tiles of 64
    const int rowBase = tileM * 128 + wave * 16;
    const int colBase = tileN6 * 64;

    // ---- async cooperative load of the B strip into LDS ----
    // 64 rows x 256 B = 1024 x 16B chunks; 256 threads x 4 chunks each.
    typedef __attribute__((address_space(3))) _Float16 lds_f16;
    const unsigned sBbase = (unsigned)(unsigned long long)(lds_f16*)sB;
#pragma unroll
    for (int r = 0; r < 4; ++r) {
        const int q   = threadIdx.x + 256 * r;
        const int row = q >> 4;
        const int c16 = q & 15;
        const unsigned goff = (unsigned)(((colBase + row) * D_ + c16 * 8) * 2);
        const unsigned loff = sBbase + (unsigned)((row * LDS_PITCH + c16 * 8) * 2);
        asm volatile("global_load_async_to_lds_b128 %0, %1, %2"
                     :: "v"(loff), "v"(goff), "s"(B) : "memory");
    }
    asm volatile("s_wait_asynccnt 0x0" ::: "memory");
    __syncthreads();

    // ---- WMMA main loop: K = 128 = 4 chunks of 32; A frag reused across 4 N tiles
    const _Float16* ap = A + (rowBase + lrow) * D_ + kbase;
    v8f acc[4] = {};
#pragma unroll
    for (int c = 0; c < 4; ++c) {
        const int kc = 32 * c;
        v8h a0 = *(const v8h*)(ap + kc);        // K = kc+kbase    .. +8
        v8h a1 = *(const v8h*)(ap + kc + 16);   // K = kc+16+kbase .. +8
        v16h av;
#pragma unroll
        for (int t = 0; t < 8; ++t) { av[t] = a0[t]; av[8 + t] = a1[t]; }
#pragma unroll
        for (int n = 0; n < 4; ++n) {
            const _Float16* bp = &sB[(n * 16 + lrow) * LDS_PITCH + kbase];
            v8h b0 = *(const v8h*)(bp + kc);
            v8h b1 = *(const v8h*)(bp + kc + 16);
            v16h bv;
#pragma unroll
            for (int t = 0; t < 8; ++t) { bv[t] = b0[t]; bv[8 + t] = b1[t]; }
            acc[n] = __builtin_amdgcn_wmma_f32_16x16x32_f16(
                /*neg_a=*/false, av, /*neg_b=*/false, bv,
                /*c_mod=*/(short)0, acc[n], /*reuse_a=*/false, /*reuse_b=*/false);
        }
    }

    // ---- epilogue. C layout: lane holds col n=lane&15; VGPR v holds row v+8*(lane>=16)
    int ct[4];
#pragma unroll
    for (int n = 0; n < 4; ++n) {
        const int j = colBase + n * 16 + lrow;
        ct[n] = (mode == 0) ? (j >> 3) : tr[j];
    }
#pragma unroll
    for (int v = 0; v < 8; ++v) {
        const int i  = rowBase + v + 8 * half_;   // same i for all 16 lanes of a half
        const int ti = tr[i];
        float esum = 0.0f;
#pragma unroll
        for (int n = 0; n < 4; ++n) {
            const float s = acc[n][v];
            if (ti == ct[n]) {
                if (mode == 0) atomicMin(&minkey[i], fkey(s));   // own-track: min candidate
            } else {
                esum += __expf(s * INV_TEMP);                    // other-track: denominator
            }
        }
        // butterfly sum across the 16 lanes sharing row i
        esum += __shfl_xor(esum, 1, 32);
        esum += __shfl_xor(esum, 2, 32);
        esum += __shfl_xor(esum, 4, 32);
        esum += __shfl_xor(esum, 8, 32);
        if (lrow == 0) atomicAdd(&den[i], esum);
    }
}

// ---- per-row finalize: num = exp(sim_p/T), den = den_y + den_x --------------
__global__ void finalize_rows(const unsigned* __restrict__ minkey,
                              const float* __restrict__ den_y,
                              const float* __restrict__ den_x,
                              float* __restrict__ num, float* __restrict__ den) {
    int i = blockIdx.x * blockDim.x + threadIdx.x;
    if (i < N_) {
        num[i] = __expf(fdec(minkey[i]) * INV_TEMP);
        den[i] = den_y[i] + den_x[i];
    }
}

// ---- loss = (1/N^2) sum_{i,j} log(1 + den[j]/num[i]) ------------------------
__global__ __launch_bounds__(256) void loss_kernel(const float* __restrict__ num,
                                                   const float* __restrict__ den,
                                                   float* __restrict__ out) {
    __shared__ float sbuf[8];
    const int i = blockIdx.x;
    const float ni = num[i];
    float p = 0.0f;
    for (int j = threadIdx.x; j < N_; j += 256)
        p += __logf(1.0f + den[j] / ni);
    p += __shfl_xor(p, 1, 32);
    p += __shfl_xor(p, 2, 32);
    p += __shfl_xor(p, 4, 32);
    p += __shfl_xor(p, 8, 32);
    p += __shfl_xor(p, 16, 32);
    if ((threadIdx.x & 31) == 0) sbuf[threadIdx.x >> 5] = p;
    __syncthreads();
    if (threadIdx.x == 0) {
        float s = 0.0f;
#pragma unroll
        for (int w = 0; w < 8; ++w) s += sbuf[w];
        atomicAdd(out, s * (1.0f / ((float)N_ * (float)N_)));
    }
}

extern "C" void kernel_launch(void* const* d_in, const int* in_sizes, int n_in,
                              void* d_out, int out_size, void* d_ws, size_t ws_size,
                              hipStream_t stream) {
    const float* x  = (const float*)d_in[0];   // [4096,128] f32
    const int*   tr = (const int*)d_in[1];     // [4096] int32
    const float* y  = (const float*)d_in[2];   // [512,8,128] f32 -> flat [4096,128]
    float* out = (float*)d_out;

    char* ws = (char*)d_ws;
    _Float16* xh     = (_Float16*)(ws);                         // 1 MB
    _Float16* yh     = (_Float16*)(ws + (1u << 20));            // 1 MB
    unsigned* minkey = (unsigned*)(ws + (2u << 20));            // 16 KB
    float*    den_y  = (float*)(ws + (2u << 20) + 1 * 16384);
    float*    den_x  = (float*)(ws + (2u << 20) + 2 * 16384);
    float*    num    = (float*)(ws + (2u << 20) + 3 * 16384);
    float*    den    = (float*)(ws + (2u << 20) + 4 * 16384);

    const int NE = N_ * D_;  // 524288

    init_kernel<<<(N_ + 255) / 256, 256, 0, stream>>>(minkey, den_y, den_x, out);
    cvt_kernel<<<(NE + 255) / 256, 256, 0, stream>>>(x, xh, NE);
    cvt_kernel<<<(NE + 255) / 256, 256, 0, stream>>>(y, yh, NE);

    // 32 macro row-tiles (128) x 64 macro col-tiles (64) -> 2048 blocks of 8 waves
    sim_kernel<<<2048, 256, 0, stream>>>(xh, yh, tr, minkey, den_y, 0);
    sim_kernel<<<2048, 256, 0, stream>>>(xh, xh, tr, minkey, den_x, 1);

    finalize_rows<<<(N_ + 255) / 256, 256, 0, stream>>>(minkey, den_y, den_x, num, den);
    loss_kernel<<<N_, 256, 0, stream>>>(num, den, out);
}
